// ceRNAnet_2963527435018
// MI455X (gfx1250) — matrix operand order repaired and checked by reading, not compiled, not run on hardware
//
#include <hip/hip_runtime.h>
#include <hip/hip_bf16.h>

// ---------------- vector types for WMMA ----------------
typedef _Float16 f16x8  __attribute__((ext_vector_type(8)));
typedef _Float16 f16x16 __attribute__((ext_vector_type(16)));
typedef float    f32x8  __attribute__((ext_vector_type(8)));

#define SAMPLES 256          // S
#define L0_BETA 0.66f
#define L0_GAMMA (-0.1f)
#define L0_ZETA 1.1f

// concatenated rna lookup: rna[s, n] with splits 8000 / 2000 / 10000
__device__ __forceinline__ float rna_at(const float* __restrict__ lnc,
                                        const float* __restrict__ mi,
                                        const float* __restrict__ mm,
                                        int s, int n) {
    if (n < 8000)  return lnc[(size_t)s * 8000 + n];
    if (n < 10000) return mi[(size_t)s * 2000 + (n - 8000)];
    return mm[(size_t)s * 10000 + (n - 10000)];
}

// ---------- K1: per-sample column stats over all N nodes (ddof=1) ----------
__global__ void col_stats_kernel(const float* __restrict__ lnc,
                                 const float* __restrict__ mi,
                                 const float* __restrict__ mm,
                                 float* __restrict__ mean,
                                 float* __restrict__ rstd,
                                 int N) {
    __shared__ float ssum[256];
    __shared__ float ssq[256];
    const int s = blockIdx.x;
    const int t = threadIdx.x;
    float acc = 0.f, acc2 = 0.f;
    for (int n = t; n < N; n += 256) {
        float v = rna_at(lnc, mi, mm, s, n);
        acc += v;
        acc2 += v * v;
    }
    ssum[t] = acc; ssq[t] = acc2;
    __syncthreads();
    for (int w = 128; w > 0; w >>= 1) {
        if (t < w) { ssum[t] += ssum[t + w]; ssq[t] += ssq[t + w]; }
        __syncthreads();
    }
    if (t == 0) {
        float mu  = ssum[0] / (float)N;
        float var = (ssq[0] - (float)N * mu * mu) / (float)(N - 1);
        mean[s] = mu;
        rstd[s] = rsqrtf(var);
    }
}

// ---------- K2: a_rna[n, s] (node-major, contiguous 1KB rows) + a1/a2 ----------
__global__ void build_arna_kernel(const float* __restrict__ lnc,
                                  const float* __restrict__ mi,
                                  const float* __restrict__ mm,
                                  const float* __restrict__ mean,
                                  const float* __restrict__ rstd,
                                  const float* __restrict__ attn_l,
                                  const float* __restrict__ attn_r,
                                  float* __restrict__ a_rna,
                                  float* __restrict__ a1,
                                  float* __restrict__ a2) {
    __shared__ float red[256];
    const int n = blockIdx.x;
    const int s = threadIdx.x;           // 256 samples
    float v = (rna_at(lnc, mi, mm, s, n) - mean[s]) * rstd[s];
    a_rna[(size_t)n * SAMPLES + s] = v;
    red[s] = v;
    __syncthreads();
    for (int w = 128; w > 0; w >>= 1) {
        if (s < w) red[s] += red[s + w];
        __syncthreads();
    }
    if (s == 0) {
        float rs = red[0];
        a1[n] = attn_l[n] * rs;
        a2[n] = attn_r[n] * rs;
    }
}

// ---------- K3: hard-concrete gate per edge + per-dst gate sum ----------
__global__ void edge_gate_kernel(const int* __restrict__ src,
                                 const int* __restrict__ dst,
                                 const float* __restrict__ a1,
                                 const float* __restrict__ a2,
                                 const float* __restrict__ bias_l0,
                                 float* __restrict__ a_e,
                                 float* __restrict__ z,
                                 int E) {
    int e = blockIdx.x * blockDim.x + threadIdx.x;
    if (e >= E) return;
    float logit = a1[src[e]] + a2[dst[e]] + bias_l0[0];
    float sg = 1.f / (1.f + __expf(-logit / L0_BETA));
    float a  = sg * (L0_ZETA - L0_GAMMA) + L0_GAMMA;
    a = fminf(fmaxf(a, 0.f), 1.f);
    a_e[e] = a;
    atomicAdd(&z[dst[e]], a);
}

// ---------- K4: edge-softmax normalization, 0/0 -> 0 ----------
__global__ void edge_norm_kernel(const int* __restrict__ dst,
                                 const float* __restrict__ z,
                                 float* __restrict__ a_e,
                                 int E) {
    int e = blockIdx.x * blockDim.x + threadIdx.x;
    if (e >= E) return;
    float zz = z[dst[e]];
    float a  = a_e[e];
    a_e[e] = (zz > 0.f) ? (a / zz) : 0.f;
}

// ---------- K5/K6: fout[dst] += a_e * fin[src], 64 threads x float4 per edge ----------
__global__ void aggregate_kernel(const int* __restrict__ src,
                                 const int* __restrict__ dst,
                                 const float* __restrict__ a_e,
                                 const float* __restrict__ fin,
                                 float* __restrict__ fout,
                                 int E) {
    int tid = blockIdx.x * blockDim.x + threadIdx.x;
    int e = tid >> 6;
    if (e >= E) return;
    float w = a_e[e];
    if (w == 0.f) return;
    int j = (tid & 63) * 4;
    const float4 v = *(const float4*)&fin[(size_t)src[e] * SAMPLES + j];
    float* o = &fout[(size_t)dst[e] * SAMPLES + j];
    atomicAdd(o + 0, w * v.x);
    atomicAdd(o + 1, w * v.y);
    atomicAdd(o + 2, w * v.z);
    atomicAdd(o + 3, w * v.w);
}

// ---------- K7: x[s, d] = (a_rna + rna1 + rna2)[idx[d], s]  (f16, K padded) ----------
__global__ void build_x_kernel(const int* __restrict__ idx,
                               const float* __restrict__ a_rna,
                               const float* __restrict__ r1,
                               const float* __restrict__ r2,
                               _Float16* __restrict__ x,
                               int D, int DP) {
    int gid = blockIdx.x * blockDim.x + threadIdx.x;
    if (gid >= SAMPLES * DP) return;
    int s = gid / DP, d = gid - s * DP;
    float v = 0.f;
    if (d < D) {
        size_t o = (size_t)idx[d] * SAMPLES + s;
        v = a_rna[o] + r1[o] + r2[o];
    }
    x[(size_t)s * DP + d] = (_Float16)v;
}

// ---------- K8/K10: pack (optionally masked) weights into padded f16 ----------
__global__ void build_w_kernel(const float* __restrict__ W,
                               const float* __restrict__ mask,   // may be null
                               _Float16* __restrict__ out,
                               int R, int C, int RP, int CP) {
    int gid = blockIdx.x * blockDim.x + threadIdx.x;
    if (gid >= RP * CP) return;
    int r = gid / CP, c = gid - r * CP;
    float v = 0.f;
    if (r < R && c < C) {
        v = W[(size_t)r * C + c];
        if (mask) v *= mask[(size_t)r * C + c];
    }
    out[(size_t)r * CP + c] = (_Float16)v;
}

// ---------- WMMA GEMM: D[m,n] = relu( sum_k A[m,k]*B[n,k] + bias[n] ) ----------
// A: [M x K] f16 row-major, B: [N x K] f16 row-major (i.e. op(B) = B^T).
// One wave = one 16x16 tile. Fragment layouts used here (wave32):
//   A: lane (mh=lane>>4, mr=lane&15): row m0+mr, halves 0..7 = K(mh*16..+7),
//      halves 8..15 = K(mh*16+8..+15)  -> two contiguous 16B loads.
//   B: lane: col n0+(lane&15), halves 0..15 = K((lane>>4)*16 .. +15)
//      -> one contiguous 32B load (K contiguous in row-major [N x K]).
//   C/D f32: lane: col n0+(lane&15), vgpr i: row m0 + i + 8*(lane>>4).
template <bool F16OUT>
__global__ void wmma_gemm_kernel(const _Float16* __restrict__ A,
                                 const _Float16* __restrict__ B,
                                 const float* __restrict__ bias,
                                 void* __restrict__ Dout,
                                 int K, int ldc, int Nlog) {
    const int lane = threadIdx.x;
    const int m0 = blockIdx.y * 16;
    const int n0 = blockIdx.x * 16;
    const int mh = lane >> 4;
    const int mr = lane & 15;

    const _Float16* arow = A + (size_t)(m0 + mr) * K;
    const _Float16* brow = B + (size_t)(n0 + mr) * K + mh * 16;

    f32x8 c = {};
    for (int k0 = 0; k0 < K; k0 += 32) {
        f16x8 alo = *(const f16x8*)(arow + k0 + mh * 16);
        f16x8 ahi = *(const f16x8*)(arow + k0 + mh * 16 + 8);
        f16x16 a, b;
        b = *(const f16x16*)(brow + k0);
#pragma unroll
        for (int i = 0; i < 8; ++i) { a[i] = alo[i]; a[i + 8] = ahi[i]; }
        c = __builtin_amdgcn_wmma_f32_16x16x32_f16(
            /*neg_a=*/false, a, /*neg_b=*/false, b,
            /*c_mod=*/(short)0, c, /*reuse_a=*/false, /*reuse_b=*/false);
    }

    const int col = n0 + mr;
    const float bs = (col < Nlog) ? bias[col] : 0.f;
#pragma unroll
    for (int i = 0; i < 8; ++i) {
        int row = m0 + i + 8 * mh;
        float v = fmaxf(c[i] + bs, 0.f);
        if (F16OUT) ((_Float16*)Dout)[(size_t)row * ldc + col] = (_Float16)v;
        else        ((float*)Dout)[(size_t)row * ldc + col]    = v;
    }
}

// ---------- K12: final 100 -> 2 projection + softmax ----------
__global__ void final_softmax_kernel(const float* __restrict__ h2,
                                     const float* __restrict__ W_po,
                                     const float* __restrict__ b_po,
                                     float* __restrict__ out,
                                     int PH, int ldh) {
    int s = blockIdx.x * blockDim.x + threadIdx.x;
    if (s >= SAMPLES) return;
    float acc0 = b_po[0], acc1 = b_po[1];
    for (int p = 0; p < PH; ++p) {
        float h = h2[(size_t)s * ldh + p];
        acc0 += h * W_po[p];
        acc1 += h * W_po[PH + p];
    }
    float mx = fmaxf(acc0, acc1);
    float e0 = __expf(acc0 - mx), e1 = __expf(acc1 - mx);
    float inv = 1.f / (e0 + e1);
    out[s * 2 + 0] = e0 * inv;
    out[s * 2 + 1] = e1 * inv;
}

extern "C" void kernel_launch(void* const* d_in, const int* in_sizes, int n_in,
                              void* d_out, int out_size, void* d_ws, size_t ws_size,
                              hipStream_t stream) {
    (void)n_in; (void)out_size; (void)ws_size;

    const float* lnc     = (const float*)d_in[0];
    const float* mi      = (const float*)d_in[1];
    const float* mm      = (const float*)d_in[2];
    const int*   src     = (const int*)d_in[3];
    const int*   dst     = (const int*)d_in[4];
    const int*   dem_idx = (const int*)d_in[5];
    const float* attn_l  = (const float*)d_in[6];
    const float* attn_r  = (const float*)d_in[7];
    const float* bias_l0 = (const float*)d_in[8];
    const float* pmask   = (const float*)d_in[9];
    const float* W_mp    = (const float*)d_in[10];
    const float* b_mp    = (const float*)d_in[11];
    const float* W_ph    = (const float*)d_in[12];
    const float* b_ph    = (const float*)d_in[13];
    const float* W_po    = (const float*)d_in[14];
    const float* b_po    = (const float*)d_in[15];

    const int E  = in_sizes[3];        // 640000
    const int N  = in_sizes[6];        // 20000
    const int D  = in_sizes[5];        // 2000
    const int P  = in_sizes[11];       // 500
    const int PH = in_sizes[13];       // 100
    const int S  = SAMPLES;            // 256

    const int KP1 = (D  + 31) & ~31;   // 2016  (K of GEMM1)
    const int PP  = (P  + 31) & ~31;   // 512   (N of GEMM1 == K of GEMM2)
    const int PHP = (PH + 15) & ~15;   // 112   (N of GEMM2)

    // ---- workspace carve-out (256B aligned) ----
    char* base = (char*)d_ws;
    size_t cur = 0;
    auto take = [&](size_t bytes) -> char* {
        cur = (cur + 255) & ~(size_t)255;
        char* p = base + cur;
        cur += bytes;
        return p;
    };
    float* mean  = (float*)take((size_t)S * 4);
    float* rstd  = (float*)take((size_t)S * 4);
    float* a_rna = (float*)take((size_t)N * S * 4);
    float* a1    = (float*)take((size_t)N * 4);
    float* a2    = (float*)take((size_t)N * 4);
    float* a_e   = (float*)take((size_t)E * 4);
    float* z     = (float*)take((size_t)N * 4);
    float* rna1  = (float*)take((size_t)N * S * 4);
    float* rna2  = (float*)take((size_t)N * S * 4);
    _Float16* x16  = (_Float16*)take((size_t)S * KP1 * 2);
    _Float16* wm16 = (_Float16*)take((size_t)PP * KP1 * 2);
    _Float16* h1   = (_Float16*)take((size_t)S * PP * 2);
    _Float16* w2   = (_Float16*)take((size_t)PHP * PP * 2);
    float*    h2   = (float*)take((size_t)S * PHP * 4);

    // ---- zero accumulators (memset nodes are graph-capturable) ----
    hipMemsetAsync(z,    0, (size_t)N * 4, stream);
    hipMemsetAsync(rna1, 0, (size_t)N * S * 4, stream);
    hipMemsetAsync(rna2, 0, (size_t)N * S * 4, stream);

    // ---- stage 1: standardize + attention sums ----
    col_stats_kernel<<<S, 256, 0, stream>>>(lnc, mi, mm, mean, rstd, N);
    build_arna_kernel<<<N, 256, 0, stream>>>(lnc, mi, mm, mean, rstd,
                                             attn_l, attn_r, a_rna, a1, a2);

    // ---- stage 2: edge gates + normalization ----
    edge_gate_kernel<<<(E + 255) / 256, 256, 0, stream>>>(src, dst, a1, a2,
                                                          bias_l0, a_e, z, E);
    edge_norm_kernel<<<(E + 255) / 256, 256, 0, stream>>>(dst, z, a_e, E);

    // ---- stage 3: two scatter-add aggregation rounds (L2-resident) ----
    unsigned agg_blocks = (unsigned)(((size_t)E * 64 + 255) / 256);
    aggregate_kernel<<<agg_blocks, 256, 0, stream>>>(src, dst, a_e, a_rna, rna1, E);
    aggregate_kernel<<<agg_blocks, 256, 0, stream>>>(src, dst, a_e, rna1, rna2, E);

    // ---- stage 4: pack f16 operands for WMMA MLP ----
    build_x_kernel<<<((S * KP1) + 255) / 256, 256, 0, stream>>>(dem_idx, a_rna, rna1,
                                                                rna2, x16, D, KP1);
    build_w_kernel<<<((PP * KP1) + 255) / 256, 256, 0, stream>>>(W_mp, pmask, wm16,
                                                                 P, D, PP, KP1);
    build_w_kernel<<<((PHP * PP) + 255) / 256, 256, 0, stream>>>(W_ph, nullptr, w2,
                                                                 PH, P, PHP, PP);

    // ---- stage 5: WMMA GEMMs (one wave per 16x16 tile) ----
    wmma_gemm_kernel<true><<<dim3(PP / 16, S / 16), 32, 0, stream>>>(
        x16, wm16, b_mp, h1, KP1, PP, P);
    wmma_gemm_kernel<false><<<dim3(PHP / 16, S / 16), 32, 0, stream>>>(
        h1, w2, b_ph, h2, PP, PHP, PH);

    // ---- stage 6: tiny output head + softmax ----
    final_softmax_kernel<<<1, 256, 0, stream>>>(h2, W_po, b_po, (float*)d_out,
                                                PH, PHP);
}
